// TorchAttractorLanguageModel_45964740001945
// MI455X (gfx1250) — matrix operand order repaired and checked by reading, not compiled, not run on hardware
//
#include <hip/hip_runtime.h>

// ---------------------------------------------------------------------------
// Problem constants (match reference)
// ---------------------------------------------------------------------------
constexpr int B_     = 1024;
constexpr int W_     = 6;
constexpr int D_     = 512;
constexpr int V_     = 50257;
constexpr int STEPS_ = 16;
constexpr int WD_    = W_ * D_;     // 3072
constexpr int ROWS_  = B_ * W_;     // 6144

constexpr float CUBGAIN_ = 0.008f * 4.0f;  // CUBIC * GAIN
constexpr float LAM_     = 0.1f;
constexpr float SIG_     = 0.5f;
constexpr float EPS_     = 1e-8f;
constexpr float LN_EPS_  = 1e-5f;
constexpr float ASYM_    = 1.25f;

// ---------------------------------------------------------------------------
// Types (wave32 CDNA5 WMMA + TDM descriptor groups)
// ---------------------------------------------------------------------------
typedef __attribute__((ext_vector_type(16))) __bf16 v16bf;
typedef __attribute__((ext_vector_type(8)))  __bf16 v8bf;
typedef __attribute__((ext_vector_type(8)))  float  v8f;
typedef __attribute__((ext_vector_type(4)))  unsigned int u32x4;
typedef __attribute__((ext_vector_type(8)))  int i32x8;
typedef __attribute__((ext_vector_type(4)))  int i32x4;

union BV { v16bf v; v8bf h[2]; };

#define DEV static __device__ __forceinline__

DEV float softplusf(float x) { return (x > 20.f) ? x : log1pf(expf(x)); }

DEV float wred(float v) {
#pragma unroll
  for (int m = 16; m >= 1; m >>= 1) v += __shfl_xor(v, m, 32);
  return v;
}

// ---------------------------------------------------------------------------
// Tensor Data Mover: async 2-D bf16 tile load Global -> LDS.
// Builds a D# per CDNA5 ISA Sec 8 (count=1, type=2 "image",
// data_size=1 -> 2 bytes, tile_dim0 x tile_dim1 with tensor_dim0_stride).
// This toolchain's builtin takes 6 args:
//   (u32x4 g0, i32x8 g1, i32x4 g2, i32x4 g3, i32x8 g4, i32 cpol)
// ---------------------------------------------------------------------------
DEV void tdm_load_2d_bf16(unsigned lds_addr, const void* gsrc,
                          unsigned tile_d0, unsigned tile_d1,
                          unsigned tensor_d0, unsigned tensor_d1,
                          unsigned stride_d0) {
  unsigned long long ga = (unsigned long long)(uintptr_t)gsrc;
  u32x4 g0;
  g0.x = 1u;                                              // count=1, user mode
  g0.y = lds_addr;                                        // lds_addr [63:32]
  g0.z = (unsigned)ga;                                    // global_addr lo
  g0.w = (unsigned)((ga >> 32) & 0x01FFFFFFu) | (2u << 30); // addr[56:32]|type=2
  i32x8 g1;
  g1[0] = 1 << 16;                                        // data_size=1 (2B)
  g1[1] = (int)((tensor_d0 & 0xFFFFu) << 16);             // tensor_dim0 lo16
  g1[2] = (int)((tensor_d0 >> 16) | ((tensor_d1 & 0xFFFFu) << 16));
  g1[3] = (int)((tensor_d1 >> 16) | (tile_d0 << 16));     // tile_dim0
  g1[4] = (int)(tile_d1 & 0xFFFFu);                       // tile_dim1, tile_dim2=0
  g1[5] = (int)stride_d0;                                 // tensor_dim0_stride lo32
  g1[6] = 0;
  g1[7] = 0;
  i32x4 gz4 = {0, 0, 0, 0};
  i32x8 gz8 = {0, 0, 0, 0, 0, 0, 0, 0};
  __builtin_amdgcn_tensor_load_to_lds(g0, g1, gz4, gz4, gz8, 0);
}

// ---------------------------------------------------------------------------
// fp32 -> bf16 conversion (grid-stride)
// ---------------------------------------------------------------------------
__global__ void f32_to_bf16(const float* __restrict__ src, __bf16* __restrict__ dst,
                            long long n) {
  long long i = (long long)blockIdx.x * blockDim.x + threadIdx.x;
  long long stride = (long long)gridDim.x * blockDim.x;
  for (; i < n; i += stride) dst[i] = (__bf16)src[i];
}

// ---------------------------------------------------------------------------
// Gather + LayerNorm: one wave32 per row of D=512 (16 floats / lane)
// ---------------------------------------------------------------------------
__global__ __launch_bounds__(256) void gather_ln(const int* __restrict__ tokens,
                                                 const float* __restrict__ embed,
                                                 float* __restrict__ S,
                                                 __bf16* __restrict__ Sbf) {
  const int lane = threadIdx.x & 31;
  const int row  = blockIdx.x * 8 + (threadIdx.x >> 5);
  if (row >= ROWS_) return;
  const int tok = tokens[row];
  const float* src = embed + (size_t)tok * D_;

  float x[16];
  float s = 0.f;
#pragma unroll
  for (int t = 0; t < 16; ++t) { x[t] = src[lane + 32 * t]; s += x[t]; }
  const float mu = wred(s) * (1.0f / D_);
  float q = 0.f;
#pragma unroll
  for (int t = 0; t < 16; ++t) { float d = x[t] - mu; q += d * d; }
  const float inv = rsqrtf(wred(q) * (1.0f / D_) + LN_EPS_);

  float* dstF = S + (size_t)row * D_;
  __bf16* dstB = Sbf + (size_t)row * D_;
#pragma unroll
  for (int t = 0; t < 16; ++t) {
    float y = (x[t] - mu) * inv;
    dstF[lane + 32 * t] = y;
    dstB[lane + 32 * t] = (__bf16)y;
  }
}

// ---------------------------------------------------------------------------
// Step matmul: MM = Sbf @ diffusion^T    ([6144 x 512] x [512 x 512])
// A tile (16 x 512 bf16, 16 KB) staged in LDS via one TENSOR_LOAD_TO_LDS per
// block; shared by all 8 waves. Each wave owns 4 N-tiles (8*4*16 = 512 cols).
// ---------------------------------------------------------------------------
__global__ __launch_bounds__(256) void step_mm(const __bf16* __restrict__ Sbf,
                                               const __bf16* __restrict__ Dbf,
                                               float* __restrict__ MM) {
  __shared__ __align__(128) __bf16 As[16 * D_];  // 16 KB

  const int lane    = threadIdx.x & 31;
  const int wave    = threadIdx.x >> 5;
  const int rowBase = blockIdx.x * 16;
  const int colBase = wave * 64;
  const int sel     = (lane < 16) ? 0 : 8;
  const int ksel    = (lane < 16) ? 0 : 16;

  if (threadIdx.x == 0)
    tdm_load_2d_bf16((unsigned)(uintptr_t)As, Sbf + (size_t)rowBase * D_,
                     /*tile*/ D_, 16, /*tensor*/ D_, ROWS_, /*stride*/ D_);
  __builtin_amdgcn_s_wait_tensorcnt(0);
  __syncthreads();

  const __bf16* Alds = As + (lane & 15) * D_;

  v8f acc[4] = {};
  for (int k0 = 0; k0 < D_; k0 += 32) {
    BV a;
    a.h[0] = *(const v8bf*)(Alds + k0 + ksel);       // ds_load_b128
    a.h[1] = *(const v8bf*)(Alds + k0 + 8 + ksel);   // ds_load_b128
    v16bf bv[4];
#pragma unroll
    for (int t = 0; t < 4; ++t) {                    // load phase
      const int col = colBase + t * 16 + (lane & 15);
      bv[t] = *(const v16bf*)(Dbf + (size_t)col * D_ + k0 + ksel);
    }
#pragma unroll
    for (int t = 0; t < 4; ++t)                      // compute phase
      acc[t] = __builtin_amdgcn_wmma_f32_16x16x32_bf16(false, a.v, false, bv[t],
                                                       (short)0, acc[t], false, false);
  }

  const int moff = (lane < 16) ? 0 : 8;
#pragma unroll
  for (int t = 0; t < 4; ++t)
#pragma unroll
    for (int r = 0; r < 8; ++r)
      MM[(size_t)(rowBase + moff + r) * D_ + colBase + t * 16 + (lane & 15)] = acc[t][r];
}

// ---------------------------------------------------------------------------
// Step elementwise update: coupling + tanh drift + normalize. One wave32/row.
// ---------------------------------------------------------------------------
__global__ __launch_bounds__(256) void step_update(const float* __restrict__ S,
                                                   const float* __restrict__ MM,
                                                   float* __restrict__ Snew,
                                                   __bf16* __restrict__ Sbf,
                                                   const float* __restrict__ pBeta,
                                                   const float* __restrict__ pGamma,
                                                   const float* __restrict__ pScale,
                                                   const float* __restrict__ pDt,
                                                   const float* __restrict__ pAsym) {
  const int lane = threadIdx.x & 31;
  const int row  = blockIdx.x * 8 + (threadIdx.x >> 5);
  if (row >= ROWS_) return;
  const int b = row / W_;
  const int i = row % W_;

  const float beta  = pBeta[0];
  const float gamma = softplusf(pGamma[0]);
  const float scale = softplusf(pScale[0]);
  const float dt    = softplusf(pDt[0]);
  const float ta    = ASYM_ * tanhf(pAsym[0]);

  float w[W_];
  float wsum = 0.f;
#pragma unroll
  for (int j = 0; j < W_; ++j) {
    float dj = (float)(j - i);
    float sg = (j > i) ? 1.f : ((j < i) ? -1.f : 0.f);
    float af = fmaxf(1.f + ta * sg, 0.2f);
    float wv = (j == i) ? 0.f : expf(-gamma * fabsf(dj)) * af;
    w[j] = wv;
    wsum += wv;
  }

  const float* Sb  = S + (size_t)b * WD_;
  const float* mmr = MM + (size_t)row * D_;

  float x[16], sig[16], mm[16];
  float ssum = 0.f;
#pragma unroll
  for (int t = 0; t < 16; ++t) {
    const int d = lane + 32 * t;
    x[t]  = Sb[i * D_ + d];
    mm[t] = mmr[d];
    float a = 0.f;
#pragma unroll
    for (int j = 0; j < W_; ++j) a += w[j] * Sb[j * D_ + d];
    sig[t] = scale * (a - wsum * x[t]);
    ssum += x[t];
  }
  const float mu = wred(ssum) * (1.0f / D_);

  float sv[16];
  float nrm2 = 0.f;
#pragma unroll
  for (int t = 0; t < 16; ++t) {
    float c     = x[t] - mu;
    float drift = mm[t] + CUBGAIN_ * tanhf(c) + beta * SIG_ * sig[t] - LAM_ * x[t];
    float s     = x[t] + dt * drift;
    if (!isfinite(s)) s = 0.f;  // nan_to_num(nan/±inf -> 0)
    sv[t] = s;
    nrm2 += s * s;
  }
  const float nrm  = sqrtf(wred(nrm2));
  const float rinv = 1.0f / (nrm + EPS_);

  float* dstF = Snew + (size_t)row * D_;
  __bf16* dstB = Sbf + (size_t)row * D_;
#pragma unroll
  for (int t = 0; t < 16; ++t) {
    float y = fminf(fmaxf(sv[t] * rinv, -10.f), 10.f);
    dstF[lane + 32 * t] = y;
    dstB[lane + 32 * t] = (__bf16)y;
  }
}

// ---------------------------------------------------------------------------
// Readout GEMM: logits[b, v] = sum_k Sbf[b, k] * Wbf[v, k]
// [1024 x 3072] x [3072 x 50257], bf16 -> f32.
// A panel (16 x 3072 bf16 = 96 KB) staged in dynamic LDS via one TDM load per
// block (320 KB/WGP -> up to 3 blocks resident). Weights stream from global;
// the 64 M-tile blocks of one N-panel launch adjacently so weight lines are
// L2-served (192 MB L2), keeping HBM weight traffic at ~309 MB total.
// ---------------------------------------------------------------------------
__global__ __launch_bounds__(256) void readout(const __bf16* __restrict__ Sbf,
                                               const __bf16* __restrict__ Wbf,
                                               float* __restrict__ out) {
  extern __shared__ __align__(128) __bf16 Apan[];  // 16 * 3072 bf16 = 96 KB

  const int lane    = threadIdx.x & 31;
  const int wave    = threadIdx.x >> 5;
  const int rowBase = blockIdx.x * 16;
  const int tile0   = (blockIdx.y * 8 + wave) * 4;
  const int sel     = (lane < 16) ? 0 : 8;
  const int ksel    = (lane < 16) ? 0 : 16;

  if (threadIdx.x == 0)
    tdm_load_2d_bf16((unsigned)(uintptr_t)Apan, Sbf + (size_t)rowBase * WD_,
                     /*tile*/ WD_, 16, /*tensor*/ WD_, B_, /*stride*/ WD_);
  __builtin_amdgcn_s_wait_tensorcnt(0);
  __syncthreads();

  const __bf16* Alds = Apan + (lane & 15) * WD_;

  int colc[4];
#pragma unroll
  for (int t = 0; t < 4; ++t) {
    int c = (tile0 + t) * 16 + (lane & 15);
    colc[t] = (c < V_) ? c : (V_ - 1);  // clamp OOB loads; stores are guarded
  }

  v8f acc[4] = {};
  for (int k0 = 0; k0 < WD_; k0 += 32) {
    BV a;
    a.h[0] = *(const v8bf*)(Alds + k0 + ksel);       // ds_load_b128
    a.h[1] = *(const v8bf*)(Alds + k0 + 8 + ksel);   // ds_load_b128
    v16bf bv[4];
#pragma unroll
    for (int t = 0; t < 4; ++t) {                    // load phase
      const __bf16* Bp = Wbf + (size_t)colc[t] * WD_ + k0 + ksel;
      bv[t] = *(const v16bf*)Bp;
      __builtin_prefetch(Bp + 128, 0, 1);            // global_prefetch_b8
    }
#pragma unroll
    for (int t = 0; t < 4; ++t)                      // compute phase
      acc[t] = __builtin_amdgcn_wmma_f32_16x16x32_bf16(false, a.v, false, bv[t],
                                                       (short)0, acc[t], false, false);
  }

  const int moff = (lane < 16) ? 0 : 8;
#pragma unroll
  for (int t = 0; t < 4; ++t) {
    const int col = (tile0 + t) * 16 + (lane & 15);
    if (col < V_) {
#pragma unroll
      for (int r = 0; r < 8; ++r)
        out[(size_t)(rowBase + moff + r) * V_ + col] = acc[t][r];
    }
  }
}

// ---------------------------------------------------------------------------
// Host launcher
// ---------------------------------------------------------------------------
extern "C" void kernel_launch(void* const* d_in, const int* in_sizes, int n_in,
                              void* d_out, int out_size, void* d_ws, size_t ws_size,
                              hipStream_t stream) {
  (void)in_sizes; (void)n_in; (void)out_size; (void)ws_size;

  const int*   tokens    = (const int*)d_in[0];
  const float* embed     = (const float*)d_in[1];
  const float* diffusion = (const float*)d_in[2];
  const float* readout_w = (const float*)d_in[3];
  const float* pBeta     = (const float*)d_in[4];
  const float* pGamma    = (const float*)d_in[5];
  const float* pScale    = (const float*)d_in[6];
  const float* pDt       = (const float*)d_in[7];
  const float* pAsym     = (const float*)d_in[8];
  float* out = (float*)d_out;

  // Workspace carve-up (16B aligned throughout)
  char* ws = (char*)d_ws;
  const size_t szS   = (size_t)ROWS_ * D_ * sizeof(float);    // 12.6 MB
  const size_t szSbf = (size_t)ROWS_ * D_ * sizeof(__bf16);   // 6.3 MB
  const size_t szDbf = (size_t)D_ * D_ * sizeof(__bf16);      // 0.5 MB
  float*  Sa   = (float*)(ws);
  float*  Sb   = (float*)(ws + szS);
  float*  MM   = (float*)(ws + 2 * szS);
  __bf16* Sbf  = (__bf16*)(ws + 3 * szS);
  __bf16* Dbf  = (__bf16*)(ws + 3 * szS + szSbf);
  __bf16* Wbf  = (__bf16*)(ws + 3 * szS + szSbf + szDbf);     // 308.8 MB

  // 1. Convert weights to bf16 (halves readout HBM traffic and unlocks the
  //    16x16x32 bf16 matrix rate -- 8x the K-depth of fp32 WMMA)
  f32_to_bf16<<<512, 256, 0, stream>>>(diffusion, Dbf, (long long)D_ * D_);
  f32_to_bf16<<<8192, 256, 0, stream>>>(readout_w, Wbf, (long long)V_ * WD_);

  // 2. Gather + LayerNorm -> S0 (fp32 + bf16 shadow)
  gather_ln<<<ROWS_ / 8, 256, 0, stream>>>(tokens, embed, Sa, Sbf);

  // 3. 16 attractor steps (TDM-staged WMMA matmul + fused elementwise update)
  float* Scur = Sa;
  float* Snxt = Sb;
  for (int s = 0; s < STEPS_; ++s) {
    step_mm<<<ROWS_ / 16, 256, 0, stream>>>(Sbf, Dbf, MM);
    step_update<<<ROWS_ / 8, 256, 0, stream>>>(Scur, MM, Snxt, Sbf,
                                               pBeta, pGamma, pScale, pDt, pAsym);
    float* t = Scur; Scur = Snxt; Snxt = t;
  }

  // 4. Readout GEMM -> logits (96 KB dynamic LDS for the TDM-staged A panel)
  constexpr int NTILES = (V_ + 15) / 16;        // 3142
  dim3 grid(B_ / 16, (NTILES + 31) / 32);       // (64, 99)
  readout<<<grid, 256, 16 * WD_ * sizeof(__bf16), stream>>>(Sbf, Wbf, out);
}
